// EquivariantInteractionBlock_56564719289126
// MI455X (gfx1250) — matrix operand order repaired and checked by reading, not compiled, not run on hardware
//
#include <hip/hip_runtime.h>

// ---------------------------------------------------------------------------
// EquivariantInteractionBlock for MI455X (gfx1250, wave32, WMMA).
//
// Roofline: dominant cost is tpw = h @ mw3 (E x 128 @ 128 x 2304 = 29.5 GFLOP).
// Materializing tpw would cost 461 MB of HBM traffic each way; instead each
// wave computes tpw 16x16-tile-at-a-time with v_wmma_f32_16x16x32_f16 and
// consumes the tile in-register (the downstream einsums are linear in tpw, and
// a 16-wide column tile maps to a single input channel of w1/w2/w3/w4).
//
// v2 changes vs round 1:
//  - 32 edges per wave (two A-fragment sets) so every mw3 B-fragment load
//    feeds two WMMAs: halves the L2->WGP B traffic per edge, which roofline
//    says was the binding term (589 KB of packed mw3 per wave sweep).
//  - silu/sigmoid/divides use v_rcp_f32 (__builtin_amdgcn_rcpf) instead of
//    IEEE divide (kills the v_div_scale/v_div_fmas sequences seen in round 1).
// ---------------------------------------------------------------------------

typedef __attribute__((ext_vector_type(16))) _Float16 v16h;
typedef __attribute__((ext_vector_type(8)))  _Float16 v8h;
typedef __attribute__((ext_vector_type(8)))  float    v8f;

#define D_TOT  80
#define N_S    32
#define N_V    16
#define HID_C  128
#define WNUM_C 2304
#define CTILES (WNUM_C/16)   // 144

static __device__ __forceinline__ float fast_rcp(float x){ return __builtin_amdgcn_rcpf(x); }
static __device__ __forceinline__ float siluf(float x){ return x * fast_rcp(1.0f + __expf(-x)); }
static __device__ __forceinline__ float sigmf(float x){ return fast_rcp(1.0f + __expf(-x)); }

static __device__ __forceinline__ v8f wmma_f16(v16h a, v16h b, v8f c){
  // D(f32 16x16) = A(f16 16x32) * B(f16 32x16) + C
  return __builtin_amdgcn_wmma_f32_16x16x32_f16(false, a, false, b, (short)0, c, false, false);
}

// A fragment (16-bit A 16x32 layout): lane holds row m=lane&15; for lane group
// g=lane>>4, element t maps to k = t + 8g + (t>=8 ? 8 : 0) -> two contiguous
// 8-half chunks at k0 = 8g and k0 = 16+8g.
static __device__ __forceinline__ v16h lds_afrag(const _Float16* p){
  v8h lo = *(const v8h*)(p);
  v8h hi = *(const v8h*)(p + 16);
  v16h a;
#pragma unroll
  for (int t = 0; t < 8; ++t){ a[t] = lo[t]; a[8+t] = hi[t]; }
  return a;
}

// ---------------------------------------------------------------------------
// Pack a row-major f32 weight (K x Ncols) into per-lane WMMA-B fragment order:
// B (f16 32x16 per tile): lane l -> n = l&15, k = kt*32 + t + 8*(l>>4) + (t>=8 ? 8 : 0).
// dst layout: [(kt*NT + nt)][lane][t] f16, i.e. 32 contiguous bytes per lane.
// ---------------------------------------------------------------------------
__global__ __launch_bounds__(256) void pack_b_kernel(const float* __restrict__ w,
                                                     _Float16* __restrict__ dst,
                                                     int K, int Ncols, int KT, int NT){
  int idx = blockIdx.x * 256 + threadIdx.x;
  int total = KT * NT * 32 * 16;
  if (idx >= total) return;
  int t    = idx & 15;
  int lane = (idx >> 4) & 31;
  int tile = idx >> 9;            // kt*NT + nt
  int nt   = tile % NT;
  int kt   = tile / NT;
  int g = lane >> 4, n = lane & 15;
  int k = kt * 32 + t + 8 * g + (t >= 8 ? 8 : 0);
  int col = nt * 16 + n;
  float v = (k < K) ? w[(size_t)k * Ncols + col] : 0.0f;
  dst[idx] = (_Float16)v;
}

// ---------------------------------------------------------------------------
// Kernel 1: irrep norm. One wave (32 lanes) per node.
// ---------------------------------------------------------------------------
__global__ __launch_bounds__(256) void norm_kernel(const float* __restrict__ x,
                                                   const float* __restrict__ nw,
                                                   const float* __restrict__ nb,
                                                   float* __restrict__ xo, int Nn){
  const int lane = threadIdx.x & 31;
  const int wave = threadIdx.x >> 5;
  const int n = blockIdx.x * 8 + wave;
  if (n >= Nn) return;
  const float* xi = x + (size_t)n * D_TOT;
  float s = xi[lane];
  float m = s;
#pragma unroll
  for (int off = 16; off >= 1; off >>= 1) m += __shfl_xor(m, off, 32);
  m *= (1.0f/32.0f);
  float dd = s - m;
  float var = dd * dd;
#pragma unroll
  for (int off = 16; off >= 1; off >>= 1) var += __shfl_xor(var, off, 32);
  var *= (1.0f/32.0f);
  float sn = dd * rsqrtf(var + 1e-8f);
  xo[(size_t)n*D_TOT + lane] = sn * nw[lane] + nb[lane];
  if (lane < 16){
    float a = xi[32+lane*3+0], b = xi[32+lane*3+1], c = xi[32+lane*3+2];
    float r = rsqrtf((a*a + b*b + c*c) * (1.0f/3.0f) + 1e-8f);
    xo[(size_t)n*D_TOT + 32 + lane*3 + 0] = a*r*nw[32+lane*3+0] + nb[32+lane*3+0];
    xo[(size_t)n*D_TOT + 32 + lane*3 + 1] = b*r*nw[32+lane*3+1] + nb[32+lane*3+1];
    xo[(size_t)n*D_TOT + 32 + lane*3 + 2] = c*r*nw[32+lane*3+2] + nb[32+lane*3+2];
  }
}

// ---------------------------------------------------------------------------
// Kernel 2: fused edge pipeline. 64 threads = 2 waves, 32 edges per wave
// (two 16-edge halves; both halves share every mw3 B-fragment load).
// ---------------------------------------------------------------------------
__global__ __launch_bounds__(64) void edge_kernel(
    const float* __restrict__ x_norm,
    const int*   __restrict__ esrc, const int* __restrict__ edst,
    const float* __restrict__ esh,  const float* __restrict__ erbf,
    const float* __restrict__ elen,
    const float* __restrict__ mb1,  const float* __restrict__ mb2,
    const float* __restrict__ mb3,
    const float* __restrict__ gb1,  const float* __restrict__ gw2,
    const float* __restrict__ gb2,
    const _Float16* __restrict__ pk_mw1, const _Float16* __restrict__ pk_mw2,
    const _Float16* __restrict__ pk_mw3, const _Float16* __restrict__ pk_gw1,
    float* __restrict__ agg, float* __restrict__ den, int E)
{
  __shared__ alignas(16) _Float16 sh_h [2][16*HID_C];   // current half: h1 then h2
  __shared__ alignas(16) _Float16 sh_rb[2][32*16];      // rbf (f16), 32 edges
  __shared__ _Float16 fA1[2][32*N_S];                   // A1*xs*sh0
  __shared__ _Float16 fA2[2][32*N_V];                   // A2*inner
  __shared__ _Float16 fA3[2][32*N_S*3];                 // A3*xs*sh1[m]
  __shared__ _Float16 fA4[2][32*N_V*3];                 // A4*xv[m]*sh0
  __shared__ float sh_ew [2][32];
  __shared__ float sh_cut[2][32];
  __shared__ int   sh_dst[2][32];

  const int lane = threadIdx.x & 31;
  const int wave = threadIdx.x >> 5;
  const int g  = lane >> 4;
  const int lr = lane & 15;
  const int eb = (blockIdx.x * 2 + wave) * 32;   // 32 edges per wave

  const float A1c = 0.125f;                 // 1/sqrt(2*NS)
  const float A2c = 0.17677669529663688f;   // 1/sqrt(2*NV)
  const float A3c = 0.125f;
  const float A4c = 0.17677669529663688f;
  const float ISQ3 = 0.5773502691896258f;

  // ---- stage 0: gather + per-edge feature staging (all 32 lanes, 1 edge each)
  {
    int ge = eb + lane;
    int valid = (ge < E);
    int gs = valid ? ge : 0;
    int src = esrc[gs];
    sh_dst[wave][lane] = edst[gs];
    float len = elen[gs];
    float cut = 0.0f;
    if (valid && len <= 5.0f)
      cut = 0.5f * (__cosf(0.6283185307179586f * len) + 1.0f);   // pi/CUT
    sh_cut[wave][lane] = cut;
#pragma unroll
    for (int k = 0; k < 16; ++k)
      sh_rb[wave][lane*16 + k] = (_Float16)(valid ? erbf[(size_t)gs*16 + k] : 0.0f);
    float s0 = esh[(size_t)gs*4 + 0];
    float s1 = esh[(size_t)gs*4 + 1];
    float s2 = esh[(size_t)gs*4 + 2];
    float s3 = esh[(size_t)gs*4 + 3];
    const float* xn = x_norm + (size_t)src * D_TOT;
#pragma unroll
    for (int i = 0; i < 32; ++i){
      float xs = xn[i];
      fA1[wave][lane*32 + i]         = (_Float16)(A1c * xs * s0);
      fA3[wave][(lane*32 + i)*3 + 0] = (_Float16)(A3c * xs * s1);
      fA3[wave][(lane*32 + i)*3 + 1] = (_Float16)(A3c * xs * s2);
      fA3[wave][(lane*32 + i)*3 + 2] = (_Float16)(A3c * xs * s3);
    }
#pragma unroll
    for (int i = 0; i < 16; ++i){
      float vx = xn[32+i*3+0], vy = xn[32+i*3+1], vz = xn[32+i*3+2];
      float inner = ISQ3 * (vx*s1 + vy*s2 + vz*s3);
      fA2[wave][lane*16 + i]         = (_Float16)(A2c * inner);
      fA4[wave][(lane*16 + i)*3 + 0] = (_Float16)(A4c * vx * s0);
      fA4[wave][(lane*16 + i)*3 + 1] = (_Float16)(A4c * vy * s0);
      fA4[wave][(lane*16 + i)*3 + 2] = (_Float16)(A4c * vz * s0);
    }
  }
  __syncthreads();

  // ---- stages 1-3 per 16-edge half: rbf -> h1 -> h2 (A frags) + edge gate
  v16h aW[2][4];   // h2 A-fragments for both halves
#pragma unroll
  for (int half = 0; half < 2; ++half){
    // A fragment of rbf (K=16 zero-padded to 32); row m=lr is edge half*16+lr
    v16h aR;
    {
      v8h lo = *(const v8h*)&sh_rb[wave][(half*16 + lr)*16 + 8*g];
#pragma unroll
      for (int t = 0; t < 8; ++t){ aR[t] = lo[t]; aR[8+t] = (_Float16)0.0f; }
    }
    // h1 = silu(rbf @ mw1 + mb1)
#pragma unroll
    for (int nt = 0; nt < 8; ++nt){
      v16h b = *(const v16h*)(pk_mw1 + ((size_t)nt*32 + lane)*16);
      v8f d = {};
      d = wmma_f16(aR, b, d);
      int c = nt*16 + lr;
      float bias = mb1[c];
#pragma unroll
      for (int v = 0; v < 8; ++v)
        sh_h[wave][(8*g + v)*HID_C + c] = (_Float16)siluf(d[v] + bias);
    }
    __syncthreads();

    // h2 = silu(h1 @ mw2 + mb2)
    v16h aH[4];
#pragma unroll
    for (int kt = 0; kt < 4; ++kt)
      aH[kt] = lds_afrag(&sh_h[wave][lr*HID_C + kt*32 + 8*g]);
    __syncthreads();
#pragma unroll
    for (int nt = 0; nt < 8; ++nt){
      v8f d = {};
#pragma unroll
      for (int kt = 0; kt < 4; ++kt){
        v16h b = *(const v16h*)(pk_mw2 + ((size_t)(kt*8 + nt)*32 + lane)*16);
        d = wmma_f16(aH[kt], b, d);
      }
      int c = nt*16 + lr;
      float bias = mb2[c];
#pragma unroll
      for (int v = 0; v < 8; ++v)
        sh_h[wave][(8*g + v)*HID_C + c] = (_Float16)siluf(d[v] + bias);
    }
    __syncthreads();

    // h2 A-fragments for the big GEMM
#pragma unroll
    for (int kt = 0; kt < 4; ++kt)
      aW[half][kt] = lds_afrag(&sh_h[wave][lr*HID_C + kt*32 + 8*g]);

    // edge gate = sigmoid(silu(rbf@gw1+gb1)@gw2+gb2) * cutoff
    float gp[8];
#pragma unroll
    for (int v = 0; v < 8; ++v) gp[v] = 0.0f;
#pragma unroll
    for (int nt = 0; nt < 8; ++nt){
      v16h b = *(const v16h*)(pk_gw1 + ((size_t)nt*32 + lane)*16);
      v8f d = {};
      d = wmma_f16(aR, b, d);
      int c = nt*16 + lr;
      float bias = gb1[c];
      float w2c  = gw2[c];
#pragma unroll
      for (int v = 0; v < 8; ++v) gp[v] += siluf(d[v] + bias) * w2c;
    }
#pragma unroll
    for (int off = 8; off >= 1; off >>= 1){
#pragma unroll
      for (int v = 0; v < 8; ++v) gp[v] += __shfl_xor(gp[v], off, 32);
    }
    if (lr == 0){
      float b2 = gb2[0];
#pragma unroll
      for (int v = 0; v < 8; ++v){
        int e32 = half*16 + 8*g + v;
        sh_ew[wave][e32] = sh_cut[wave][e32] * sigmf(gp[v] + b2);
      }
    }
    __syncthreads();   // sh_h free for the next half
  }

  // ---- stage 4: tpw = h2 @ mw3 + mb3, consumed tile-by-tile (both halves
  // share each B-fragment load)
  float aS0[2][8], aS1[2][8], aV0[2][8], aV1[2][8], aV2[2][8];
#pragma unroll
  for (int h = 0; h < 2; ++h)
#pragma unroll
    for (int v = 0; v < 8; ++v){
      aS0[h][v]=0.f; aS1[h][v]=0.f; aV0[h][v]=0.f; aV1[h][v]=0.f; aV2[h][v]=0.f;
    }

  // segment 1: w1 (NS x NS), cols [0,1024): tile ct -> i = ct>>1, o-half = ct&1
  for (int ct = 0; ct < 64; ++ct){
    v8f d0 = {}, d1 = {};
#pragma unroll
    for (int kt = 0; kt < 4; ++kt){
      v16h b = *(const v16h*)(pk_mw3 + ((size_t)(kt*CTILES + ct)*32 + lane)*16);
      d0 = wmma_f16(aW[0][kt], b, d0);
      d1 = wmma_f16(aW[1][kt], b, d1);
    }
    float b3 = mb3[ct*16 + lr];
    int i = ct >> 1;
    if ((ct & 1) == 0){
#pragma unroll
      for (int v = 0; v < 8; ++v){
        aS0[0][v] += (float)fA1[wave][(     8*g+v)*32 + i] * (d0[v] + b3);
        aS0[1][v] += (float)fA1[wave][(16 + 8*g+v)*32 + i] * (d1[v] + b3);
      }
    } else {
#pragma unroll
      for (int v = 0; v < 8; ++v){
        aS1[0][v] += (float)fA1[wave][(     8*g+v)*32 + i] * (d0[v] + b3);
        aS1[1][v] += (float)fA1[wave][(16 + 8*g+v)*32 + i] * (d1[v] + b3);
      }
    }
  }
  // segment 2: w2 (NV x NS), cols [1024,1536)
  for (int ct = 64; ct < 96; ++ct){
    v8f d0 = {}, d1 = {};
#pragma unroll
    for (int kt = 0; kt < 4; ++kt){
      v16h b = *(const v16h*)(pk_mw3 + ((size_t)(kt*CTILES + ct)*32 + lane)*16);
      d0 = wmma_f16(aW[0][kt], b, d0);
      d1 = wmma_f16(aW[1][kt], b, d1);
    }
    float b3 = mb3[ct*16 + lr];
    int cc = ct - 64;
    int i = cc >> 1;
    if ((cc & 1) == 0){
#pragma unroll
      for (int v = 0; v < 8; ++v){
        aS0[0][v] += (float)fA2[wave][(     8*g+v)*16 + i] * (d0[v] + b3);
        aS0[1][v] += (float)fA2[wave][(16 + 8*g+v)*16 + i] * (d1[v] + b3);
      }
    } else {
#pragma unroll
      for (int v = 0; v < 8; ++v){
        aS1[0][v] += (float)fA2[wave][(     8*g+v)*16 + i] * (d0[v] + b3);
        aS1[1][v] += (float)fA2[wave][(16 + 8*g+v)*16 + i] * (d1[v] + b3);
      }
    }
  }
  // segment 3: w3 (NS x NV), cols [1536,2048): tile ct -> i = ct-96, o = lr
  for (int ct = 96; ct < 128; ++ct){
    v8f d0 = {}, d1 = {};
#pragma unroll
    for (int kt = 0; kt < 4; ++kt){
      v16h b = *(const v16h*)(pk_mw3 + ((size_t)(kt*CTILES + ct)*32 + lane)*16);
      d0 = wmma_f16(aW[0][kt], b, d0);
      d1 = wmma_f16(aW[1][kt], b, d1);
    }
    float b3 = mb3[ct*16 + lr];
    int i = ct - 96;
#pragma unroll
    for (int v = 0; v < 8; ++v){
      int e0 = 8*g + v, e1 = 16 + 8*g + v;
      float dv0 = d0[v] + b3, dv1 = d1[v] + b3;
      aV0[0][v] += (float)fA3[wave][(e0*32 + i)*3 + 0] * dv0;
      aV1[0][v] += (float)fA3[wave][(e0*32 + i)*3 + 1] * dv0;
      aV2[0][v] += (float)fA3[wave][(e0*32 + i)*3 + 2] * dv0;
      aV0[1][v] += (float)fA3[wave][(e1*32 + i)*3 + 0] * dv1;
      aV1[1][v] += (float)fA3[wave][(e1*32 + i)*3 + 1] * dv1;
      aV2[1][v] += (float)fA3[wave][(e1*32 + i)*3 + 2] * dv1;
    }
  }
  // segment 4: w4 (NV x NV), cols [2048,2304): tile ct -> i = ct-128, o = lr
  for (int ct = 128; ct < 144; ++ct){
    v8f d0 = {}, d1 = {};
#pragma unroll
    for (int kt = 0; kt < 4; ++kt){
      v16h b = *(const v16h*)(pk_mw3 + ((size_t)(kt*CTILES + ct)*32 + lane)*16);
      d0 = wmma_f16(aW[0][kt], b, d0);
      d1 = wmma_f16(aW[1][kt], b, d1);
    }
    float b3 = mb3[ct*16 + lr];
    int i = ct - 128;
#pragma unroll
    for (int v = 0; v < 8; ++v){
      int e0 = 8*g + v, e1 = 16 + 8*g + v;
      float dv0 = d0[v] + b3, dv1 = d1[v] + b3;
      aV0[0][v] += (float)fA4[wave][(e0*16 + i)*3 + 0] * dv0;
      aV1[0][v] += (float)fA4[wave][(e0*16 + i)*3 + 1] * dv0;
      aV2[0][v] += (float)fA4[wave][(e0*16 + i)*3 + 2] * dv0;
      aV0[1][v] += (float)fA4[wave][(e1*16 + i)*3 + 0] * dv1;
      aV1[1][v] += (float)fA4[wave][(e1*16 + i)*3 + 1] * dv1;
      aV2[1][v] += (float)fA4[wave][(e1*16 + i)*3 + 2] * dv1;
    }
  }

  // ---- scatter: m *= edge_w; segment-sum into agg / den (edge_w==0 for tail)
#pragma unroll
  for (int h = 0; h < 2; ++h){
#pragma unroll
    for (int v = 0; v < 8; ++v){
      int e32 = h*16 + 8*g + v;
      float w = sh_ew[wave][e32];
      int dst = sh_dst[wave][e32];
      float* base = agg + (size_t)dst * D_TOT;
      atomicAdd(base + lr,              aS0[h][v]*w);
      atomicAdd(base + 16 + lr,         aS1[h][v]*w);
      atomicAdd(base + 32 + lr*3 + 0,   aV0[h][v]*w);
      atomicAdd(base + 32 + lr*3 + 1,   aV1[h][v]*w);
      atomicAdd(base + 32 + lr*3 + 2,   aV2[h][v]*w);
      if (lr == ((8*g + v) & 15)) atomicAdd(den + dst, w);
    }
  }
}

// ---------------------------------------------------------------------------
// Kernel 3: node update. One wave per node; small matmuls via VALU.
// ---------------------------------------------------------------------------
__global__ __launch_bounds__(256) void node_kernel(
    const float* __restrict__ x,   const float* __restrict__ xnorm,
    const float* __restrict__ agg, const float* __restrict__ den,
    const float* __restrict__ Ws,  const float* __restrict__ Wv,
    const float* __restrict__ Us,  const float* __restrict__ Uv,
    const float* __restrict__ Ss,  const float* __restrict__ Sv,
    const float* __restrict__ res_scale, float* __restrict__ out, int Nn)
{
  __shared__ float lds[8][3][D_TOT];
  const int lane = threadIdx.x & 31;
  const int wave = threadIdx.x >> 5;
  const int n = blockIdx.x * 8 + wave;
  if (n >= Nn) return;
  float* ag = lds[wave][0];
  float* xn = lds[wave][1];
  float* gv = lds[wave][2];
  const float rdn = fast_rcp(fmaxf(den[n], 1e-8f));
  for (int j = lane; j < D_TOT; j += 32){
    ag[j] = agg[(size_t)n*D_TOT + j] * rdn;
    xn[j] = xnorm[(size_t)n*D_TOT + j];
  }
  __builtin_amdgcn_wave_barrier();
  const float rs   = res_scale[0];
  const float is32 = 0.17677669529663688f;  // 1/sqrt(NS)
  const float is16 = 0.25f;                 // 1/sqrt(NV)
  // sc = ag_s @ Ws / sqrt(NS): lane j<32 -> col j; lanes 0..15 also col 32+j
  float sa = 0.f, sb = 0.f;
  for (int i = 0; i < 32; ++i){
    float a = ag[i];
    sa += a * Ws[i*48 + lane];
    sb += a * Ws[i*48 + 32 + (lane & 15)];
  }
  float s_g  = siluf(sa * is32);
  float gate = sigmf(sb * is32);
  float v0 = 0.f, v1 = 0.f, v2 = 0.f;
  if (lane < 16){
    for (int i = 0; i < 16; ++i){
      float w = Wv[i*16 + lane];
      v0 += ag[32+i*3+0] * w;
      v1 += ag[32+i*3+1] * w;
      v2 += ag[32+i*3+2] * w;
    }
    v0 *= is16 * gate; v1 *= is16 * gate; v2 *= is16 * gate;
  }
  gv[lane] = s_g;
  if (lane < 16){ gv[32+lane*3+0]=v0; gv[32+lane*3+1]=v1; gv[32+lane*3+2]=v2; }
  __builtin_amdgcn_wave_barrier();
  // upd_s + self_s
  float os = 0.f;
  for (int i = 0; i < 32; ++i)
    os += gv[i]*Us[i*32 + lane] + xn[i]*Ss[i*32 + lane];
  os *= is32;
  out[(size_t)n*D_TOT + lane] = x[(size_t)n*D_TOT + lane] + rs*os;
  if (lane < 16){
    float o0 = 0.f, o1 = 0.f, o2 = 0.f;
    for (int i = 0; i < 16; ++i){
      float u = Uv[i*16 + lane], s = Sv[i*16 + lane];
      o0 += gv[32+i*3+0]*u + xn[32+i*3+0]*s;
      o1 += gv[32+i*3+1]*u + xn[32+i*3+1]*s;
      o2 += gv[32+i*3+2]*u + xn[32+i*3+2]*s;
    }
    out[(size_t)n*D_TOT + 32 + lane*3 + 0] = x[(size_t)n*D_TOT + 32 + lane*3 + 0] + rs*(o0*is16);
    out[(size_t)n*D_TOT + 32 + lane*3 + 1] = x[(size_t)n*D_TOT + 32 + lane*3 + 1] + rs*(o1*is16);
    out[(size_t)n*D_TOT + 32 + lane*3 + 2] = x[(size_t)n*D_TOT + 32 + lane*3 + 2] + rs*(o2*is16);
  }
}

// ---------------------------------------------------------------------------
extern "C" void kernel_launch(void* const* d_in, const int* in_sizes, int n_in,
                              void* d_out, int out_size, void* d_ws, size_t ws_size,
                              hipStream_t stream)
{
  (void)n_in; (void)out_size; (void)ws_size;
  const float* x      = (const float*)d_in[0];
  const int*   esrc   = (const int*)  d_in[1];
  const int*   edst   = (const int*)  d_in[2];
  const float* esh    = (const float*)d_in[3];
  const float* erbf   = (const float*)d_in[4];
  const float* elen   = (const float*)d_in[5];
  const float* norm_w = (const float*)d_in[6];
  const float* norm_b = (const float*)d_in[7];
  const float* mw1    = (const float*)d_in[8];
  const float* mb1    = (const float*)d_in[9];
  const float* mw2    = (const float*)d_in[10];
  const float* mb2    = (const float*)d_in[11];
  const float* mw3    = (const float*)d_in[12];
  const float* mb3    = (const float*)d_in[13];
  const float* gw1    = (const float*)d_in[14];
  const float* gb1    = (const float*)d_in[15];
  const float* gw2    = (const float*)d_in[16];
  const float* gb2    = (const float*)d_in[17];
  const float* Ws     = (const float*)d_in[18];
  const float* Wv     = (const float*)d_in[19];
  const float* Us     = (const float*)d_in[20];
  const float* Uv     = (const float*)d_in[21];
  const float* Ss     = (const float*)d_in[22];
  const float* Sv     = (const float*)d_in[23];
  const float* res    = (const float*)d_in[24];

  const int Nn = in_sizes[0] / D_TOT;
  const int E  = in_sizes[1];

  char* ws = (char*)d_ws;
  size_t off = 0;
  auto take = [&](size_t bytes) -> char* {
    char* p = ws + off;
    off = (off + bytes + 255) & ~(size_t)255;
    return p;
  };
  float*     agg    = (float*)    take((size_t)Nn * D_TOT * 4);
  float*     den    = (float*)    take((size_t)Nn * 4);
  float*     xnorm  = (float*)    take((size_t)Nn * D_TOT * 4);
  _Float16*  pk_mw1 = (_Float16*) take((size_t)1 * 8      * 512 * 2);
  _Float16*  pk_gw1 = (_Float16*) take((size_t)1 * 8      * 512 * 2);
  _Float16*  pk_mw2 = (_Float16*) take((size_t)4 * 8      * 512 * 2);
  _Float16*  pk_mw3 = (_Float16*) take((size_t)4 * CTILES * 512 * 2);

  hipMemsetAsync(agg, 0, (size_t)Nn * D_TOT * 4, stream);
  hipMemsetAsync(den, 0, (size_t)Nn * 4, stream);

  { int tot = 1*8*512;      pack_b_kernel<<<(tot+255)/256, 256, 0, stream>>>(mw1, pk_mw1, 16,  HID_C,  1, 8); }
  { int tot = 1*8*512;      pack_b_kernel<<<(tot+255)/256, 256, 0, stream>>>(gw1, pk_gw1, 16,  HID_C,  1, 8); }
  { int tot = 4*8*512;      pack_b_kernel<<<(tot+255)/256, 256, 0, stream>>>(mw2, pk_mw2, 128, HID_C,  4, 8); }
  { int tot = 4*CTILES*512; pack_b_kernel<<<(tot+255)/256, 256, 0, stream>>>(mw3, pk_mw3, 128, WNUM_C, 4, CTILES); }

  norm_kernel<<<(Nn + 7)/8, 256, 0, stream>>>(x, norm_w, norm_b, xnorm, Nn);

  edge_kernel<<<(E + 63)/64, 64, 0, stream>>>(xnorm, esrc, edst, esh, erbf, elen,
                                              mb1, mb2, mb3, gb1, gw2, gb2,
                                              pk_mw1, pk_mw2, pk_mw3, pk_gw1,
                                              agg, den, E);

  node_kernel<<<(Nn + 7)/8, 256, 0, stream>>>(x, xnorm, agg, den,
                                              Ws, Wv, Us, Uv, Ss, Sv,
                                              res, (float*)d_out, Nn);
}
